// MultiheadedSelfAttention_2422361555213
// MI455X (gfx1250) — compile-verified, matrix operand order checked by reading, not run on hardware
//
#include <hip/hip_runtime.h>
#include <hip/hip_bf16.h>

// ---------------------------------------------------------------------------
// MI455X (gfx1250) multi-head self-attention.
// Compute-bound (~103 GFLOP vs ~84 MB HBM traffic) -> bf16 WMMA with f32
// accumulation (v_wmma_f32_16x16x32_bf16) is the rewarded path on CDNA5.
// ---------------------------------------------------------------------------

typedef __attribute__((ext_vector_type(16))) __bf16          bf16x16;
typedef __attribute__((ext_vector_type(8)))  __bf16          bf16x8;
typedef __attribute__((ext_vector_type(8)))  float           f32x8;
typedef __attribute__((ext_vector_type(4)))  float           f32x4;
typedef __attribute__((ext_vector_type(8)))  unsigned short  u16x8;
typedef __attribute__((ext_vector_type(4)))  unsigned short  u16x4;

union BF16x16U { bf16x16 v; bf16x8 h[2]; };

static constexpr int Bq = 8, Nq = 1024, Cq = 1024, Hq = 16, HD = 64;
static constexpr int M_ROWS = Bq * Nq;   // 8192 token rows
static constexpr int QKV_D  = 3 * Cq;    // 3072 output features of QKV proj

__device__ __forceinline__ unsigned short f2bf(float f) {
  unsigned int u = __builtin_bit_cast(unsigned int, f);
  u += 0x7FFFu + ((u >> 16) & 1u);       // round-to-nearest-even
  return (unsigned short)(u >> 16);
}

__device__ __forceinline__ f32x8 wmma_bf16(bf16x16 a, bf16x16 b, f32x8 c) {
  // (neg_a, A, neg_b, B, c_mod, C, reuse_a, reuse_b)
  return __builtin_amdgcn_wmma_f32_16x16x32_bf16(false, a, false, b,
                                                 (short)0, c, false, false);
}

__device__ __forceinline__ float rmax16(float v) {
  v = fmaxf(v, __shfl_xor(v, 1));
  v = fmaxf(v, __shfl_xor(v, 2));
  v = fmaxf(v, __shfl_xor(v, 4));
  v = fmaxf(v, __shfl_xor(v, 8));
  return v;  // reduced across the 16-lane group (matches C-matrix striping)
}
__device__ __forceinline__ float rsum16(float v) {
  v += __shfl_xor(v, 1);
  v += __shfl_xor(v, 2);
  v += __shfl_xor(v, 4);
  v += __shfl_xor(v, 8);
  return v;
}

// --------------------------- stage 0: fp32 -> bf16 -------------------------
__global__ void cvt_f32_bf16(const float* __restrict__ in,
                             unsigned short* __restrict__ out, int n4) {
  int i = blockIdx.x * blockDim.x + threadIdx.x;
  if (i < n4) {
    f32x4 v = ((const f32x4*)in)[i];
    u16x4 o;
    o.x = f2bf(v.x); o.y = f2bf(v.y); o.z = f2bf(v.z); o.w = f2bf(v.w);
    ((u16x4*)out)[i] = o;
  }
}

// --------------------------- stage 1: QKV projection ------------------------
// qkv[m,d] = sum_c x[m,c] * w_qkv[d,c] + b_qkv[d]
// Wave computes a 32x64 tile (2x4 accumulators). Both operands K-contiguous.
// Epilogue: Q,K -> (B,H,N,hd) bf16 (Q pre-scaled by hd^-0.5), V -> (B,H,hd,N).
__global__ __launch_bounds__(128)
void qkv_gemm(const unsigned short* __restrict__ Xb,
              const unsigned short* __restrict__ Wb,
              const float* __restrict__ bias,
              unsigned short* __restrict__ Q,
              unsigned short* __restrict__ Km,
              unsigned short* __restrict__ VT) {
  const int lane = threadIdx.x & 31;
  const int half = lane >> 4;
  const int l16  = lane & 15;
  const int wave = blockIdx.x * (blockDim.x >> 5) + (threadIdx.x >> 5);
  const int tiles_m = M_ROWS / 32;          // 256
  const int m0 = (wave % tiles_m) * 32;
  const int n0 = (wave / tiles_m) * 64;     // 48 column tiles

  f32x8 acc[2][4];
#pragma unroll
  for (int i = 0; i < 2; ++i)
#pragma unroll
    for (int j = 0; j < 4; ++j) acc[i][j] = (f32x8)0.0f;

  const unsigned short* arow[2];
  arow[0] = Xb + (size_t)(m0 + l16) * Cq;
  arow[1] = Xb + (size_t)(m0 + 16 + l16) * Cq;
  const unsigned short* brow[4];
#pragma unroll
  for (int j = 0; j < 4; ++j) brow[j] = Wb + (size_t)(n0 + 16 * j + l16) * Cq;

  for (int k0 = 0; k0 < Cq; k0 += 32) {
    const int o0 = k0 + half * 8;
    const int o1 = k0 + 16 + half * 8;
    BF16x16U a[2], b[4];
#pragma unroll
    for (int i = 0; i < 2; ++i) {
      a[i].h[0] = *(const bf16x8*)(arow[i] + o0);
      a[i].h[1] = *(const bf16x8*)(arow[i] + o1);
    }
#pragma unroll
    for (int j = 0; j < 4; ++j) {
      b[j].h[0] = *(const bf16x8*)(brow[j] + o0);
      b[j].h[1] = *(const bf16x8*)(brow[j] + o1);
    }
#pragma unroll
    for (int i = 0; i < 2; ++i)
#pragma unroll
      for (int j = 0; j < 4; ++j)
        acc[i][j] = wmma_bf16(a[i].v, b[j].v, acc[i][j]);
  }

  const int bidx = m0 >> 10;                // batch (1024 rows per batch)
#pragma unroll
  for (int j = 0; j < 4; ++j) {
    const int d   = n0 + 16 * j + l16;      // this lane's output feature
    const float bv = bias[d];
    const int seg = d >> 10;                // 0:q 1:k 2:v
    const int dl  = d & 1023;
    const int hh  = dl >> 6;
    const int hd  = dl & 63;
#pragma unroll
    for (int i = 0; i < 2; ++i) {
      const int nbase = (m0 & 1023) + 16 * i + half * 8;   // 8 consecutive rows
      if (seg == 2) {                       // V: transposed store, contiguous n
        u16x8 pk;
#pragma unroll
        for (int r = 0; r < 8; ++r) pk[r] = f2bf(acc[i][j][r] + bv);
        *(u16x8*)(VT + ((size_t)(bidx * Hq + hh) * HD + hd) * Nq + nbase) = pk;
      } else {
        unsigned short* base =
            (seg == 0 ? Q : Km) + (size_t)(bidx * Hq + hh) * Nq * HD + hd;
        const float sc = (seg == 0) ? 0.125f : 1.0f;  // hd^-0.5 folded into Q
#pragma unroll
        for (int r = 0; r < 8; ++r)
          base[(size_t)(nbase + r) * HD] = f2bf((acc[i][j][r] + bv) * sc);
      }
    }
  }
}

// --------------------------- stage 2: flash attention -----------------------
// One wave owns 16 query rows of one (b,h); streams keys in blocks of 32 with
// online softmax. P goes through LDS to re-stripe C-layout -> A-layout.
__global__ __launch_bounds__(128)
void attn_kernel(const unsigned short* __restrict__ Q,
                 const unsigned short* __restrict__ Km,
                 const unsigned short* __restrict__ VT,
                 unsigned short* __restrict__ AO) {
  __shared__ unsigned short Pl[4][16 * 32];
  const int lane = threadIdx.x & 31;
  const int half = lane >> 4;
  const int l16  = lane & 15;
  const int wv   = threadIdx.x >> 5;
  const int wave = blockIdx.x * 4 + wv;     // 8192 waves total
  const int b  = wave >> 10;
  const int hh = (wave >> 6) & 15;
  const int q0 = (wave & 63) * 16;

  const unsigned short* qbase = Q  + (size_t)(b * Hq + hh) * Nq * HD;
  const unsigned short* kbase = Km + (size_t)(b * Hq + hh) * Nq * HD;
  const unsigned short* vbase = VT + (size_t)(b * Hq + hh) * HD * Nq;

  // Q rows as A operands for c=0..31 and c=32..63
  BF16x16U aq0, aq1;
  {
    const unsigned short* qr = qbase + (size_t)(q0 + l16) * HD;
    aq0.h[0] = *(const bf16x8*)(qr + half * 8);
    aq0.h[1] = *(const bf16x8*)(qr + 16 + half * 8);
    aq1.h[0] = *(const bf16x8*)(qr + 32 + half * 8);
    aq1.h[1] = *(const bf16x8*)(qr + 48 + half * 8);
  }

  float rmax[8], rsum[8];
  f32x8 o[4];
#pragma unroll
  for (int r = 0; r < 8; ++r) { rmax[r] = -1e30f; rsum[r] = 0.0f; }
#pragma unroll
  for (int j = 0; j < 4; ++j) o[j] = (f32x8)0.0f;

  unsigned short* pl = &Pl[wv][0];

  for (int kt = 0; kt < Nq; kt += 32) {
    // S = Q.K^T for 32 keys (two 16-key column tiles)
    f32x8 s0 = (f32x8)0.0f, s1 = (f32x8)0.0f;
    {
      BF16x16U bk0, bk1;
      const unsigned short* kr = kbase + (size_t)(kt + l16) * HD;
      bk0.h[0] = *(const bf16x8*)(kr + half * 8);
      bk0.h[1] = *(const bf16x8*)(kr + 16 + half * 8);
      bk1.h[0] = *(const bf16x8*)(kr + 32 + half * 8);
      bk1.h[1] = *(const bf16x8*)(kr + 48 + half * 8);
      s0 = wmma_bf16(aq0.v, bk0.v, s0);
      s0 = wmma_bf16(aq1.v, bk1.v, s0);
      const unsigned short* kr2 = kbase + (size_t)(kt + 16 + l16) * HD;
      bk0.h[0] = *(const bf16x8*)(kr2 + half * 8);
      bk0.h[1] = *(const bf16x8*)(kr2 + 16 + half * 8);
      bk1.h[0] = *(const bf16x8*)(kr2 + 32 + half * 8);
      bk1.h[1] = *(const bf16x8*)(kr2 + 48 + half * 8);
      s1 = wmma_bf16(aq0.v, bk0.v, s1);
      s1 = wmma_bf16(aq1.v, bk1.v, s1);
    }

    // online softmax over this 32-key block
    float corr[8];
#pragma unroll
    for (int r = 0; r < 8; ++r) {
      float tmax = rmax16(fmaxf(s0[r], s1[r]));
      float nm   = fmaxf(rmax[r], tmax);
      corr[r]    = __expf(rmax[r] - nm);
      rmax[r]    = nm;
      float p0 = __expf(s0[r] - nm);
      float p1 = __expf(s1[r] - nm);
      rsum[r] = rsum[r] * corr[r] + rsum16(p0 + p1);
      s0[r] = p0; s1[r] = p1;
    }
#pragma unroll
    for (int j = 0; j < 4; ++j)
#pragma unroll
      for (int r = 0; r < 8; ++r) o[j][r] *= corr[r];

    // C-layout P -> LDS (row-major 16x32 bf16), then reload in A-layout
#pragma unroll
    for (int r = 0; r < 8; ++r) {
      pl[(r + 8 * half) * 32 + l16]      = f2bf(s0[r]);
      pl[(r + 8 * half) * 32 + 16 + l16] = f2bf(s1[r]);
    }
    asm volatile("s_wait_dscnt 0" ::: "memory");
    BF16x16U ap;
    ap.h[0] = *(const bf16x8*)(pl + l16 * 32 + half * 8);
    ap.h[1] = *(const bf16x8*)(pl + l16 * 32 + 16 + half * 8);

    // O += P.V  (V stored transposed: contiguous in key index)
#pragma unroll
    for (int j = 0; j < 4; ++j) {
      BF16x16U bv;
      const unsigned short* vr = vbase + (size_t)(j * 16 + l16) * Nq + kt;
      bv.h[0] = *(const bf16x8*)(vr + half * 8);
      bv.h[1] = *(const bf16x8*)(vr + 16 + half * 8);
      o[j] = wmma_bf16(ap.v, bv.v, o[j]);
    }
  }

  // normalize and write attention output as (B,N,C) bf16
#pragma unroll
  for (int r = 0; r < 8; ++r) rsum[r] = 1.0f / rsum[r];
#pragma unroll
  for (int j = 0; j < 4; ++j) {
#pragma unroll
    for (int r = 0; r < 8; ++r) {
      const int n = q0 + r + 8 * half;
      AO[(size_t)(b * Nq + n) * Cq + hh * HD + j * 16 + l16] =
          f2bf(o[j][r] * rsum[r]);
    }
  }
}

// --------------------------- stage 3: output projection ---------------------
__global__ __launch_bounds__(128)
void proj_gemm(const unsigned short* __restrict__ Ab,
               const unsigned short* __restrict__ Wb,
               const float* __restrict__ bias,
               float* __restrict__ Out) {
  const int lane = threadIdx.x & 31;
  const int half = lane >> 4;
  const int l16  = lane & 15;
  const int wave = blockIdx.x * (blockDim.x >> 5) + (threadIdx.x >> 5);
  const int tiles_m = M_ROWS / 32;          // 256
  const int m0 = (wave % tiles_m) * 32;
  const int n0 = (wave / tiles_m) * 64;     // 16 column tiles

  f32x8 acc[2][4];
#pragma unroll
  for (int i = 0; i < 2; ++i)
#pragma unroll
    for (int j = 0; j < 4; ++j) acc[i][j] = (f32x8)0.0f;

  const unsigned short* arow[2];
  arow[0] = Ab + (size_t)(m0 + l16) * Cq;
  arow[1] = Ab + (size_t)(m0 + 16 + l16) * Cq;
  const unsigned short* brow[4];
#pragma unroll
  for (int j = 0; j < 4; ++j) brow[j] = Wb + (size_t)(n0 + 16 * j + l16) * Cq;

  for (int k0 = 0; k0 < Cq; k0 += 32) {
    const int o0 = k0 + half * 8;
    const int o1 = k0 + 16 + half * 8;
    BF16x16U a[2], b[4];
#pragma unroll
    for (int i = 0; i < 2; ++i) {
      a[i].h[0] = *(const bf16x8*)(arow[i] + o0);
      a[i].h[1] = *(const bf16x8*)(arow[i] + o1);
    }
#pragma unroll
    for (int j = 0; j < 4; ++j) {
      b[j].h[0] = *(const bf16x8*)(brow[j] + o0);
      b[j].h[1] = *(const bf16x8*)(brow[j] + o1);
    }
#pragma unroll
    for (int i = 0; i < 2; ++i)
#pragma unroll
      for (int j = 0; j < 4; ++j)
        acc[i][j] = wmma_bf16(a[i].v, b[j].v, acc[i][j]);
  }

#pragma unroll
  for (int j = 0; j < 4; ++j) {
    const int d = n0 + 16 * j + l16;
    const float bv = bias[d];
#pragma unroll
    for (int i = 0; i < 2; ++i) {
      const int rb = m0 + 16 * i + half * 8;
#pragma unroll
      for (int r = 0; r < 8; ++r)
        Out[(size_t)(rb + r) * Cq + d] = acc[i][j][r] + bv;
    }
  }
}

// ---------------------------------------------------------------------------
extern "C" void kernel_launch(void* const* d_in, const int* in_sizes, int n_in,
                              void* d_out, int out_size, void* d_ws,
                              size_t ws_size, hipStream_t stream) {
  const float* x      = (const float*)d_in[0];  // (8,1024,1024)
  const float* w_qkv  = (const float*)d_in[1];  // (3072,1024)
  const float* b_qkv  = (const float*)d_in[2];  // (3072,)
  const float* w_proj = (const float*)d_in[3];  // (1024,1024)
  const float* b_proj = (const float*)d_in[4];  // (1024,)
  float* out = (float*)d_out;

  unsigned short* ws = (unsigned short*)d_ws;   // offsets in u16 elements
  unsigned short* XB    = ws;                       // 8,388,608
  unsigned short* WQKVB = XB    + (size_t)M_ROWS * Cq;   // 3,145,728
  unsigned short* WPRJB = WQKVB + (size_t)QKV_D * Cq;    // 1,048,576
  unsigned short* Qb    = WPRJB + (size_t)Cq * Cq;       // 8,388,608
  unsigned short* Kb    = Qb    + (size_t)M_ROWS * Cq;
  unsigned short* VTb   = Kb    + (size_t)M_ROWS * Cq;
  unsigned short* AOb   = VTb   + (size_t)M_ROWS * Cq;   // total ~92.3 MB

  // stage 0: convert operands to bf16
  {
    int n4 = (M_ROWS * Cq) / 4;
    cvt_f32_bf16<<<n4 / 256, 256, 0, stream>>>(x, XB, n4);
    n4 = (QKV_D * Cq) / 4;
    cvt_f32_bf16<<<n4 / 256, 256, 0, stream>>>(w_qkv, WQKVB, n4);
    n4 = (Cq * Cq) / 4;
    cvt_f32_bf16<<<n4 / 256, 256, 0, stream>>>(w_proj, WPRJB, n4);
  }

  // stage 1: QKV projection -> Q,K (B,H,N,hd), V transposed (B,H,hd,N)
  {
    const int waves = (M_ROWS / 32) * (QKV_D / 64);  // 12288
    qkv_gemm<<<waves / 4, 128, 0, stream>>>(XB, WQKVB, b_qkv, Qb, Kb, VTb);
  }

  // stage 2: flash attention -> AO (B,N,C) bf16
  {
    const int waves = Bq * Hq * (Nq / 16);           // 8192
    attn_kernel<<<waves / 4, 128, 0, stream>>>(Qb, Kb, VTb, AOb);
  }

  // stage 3: output projection -> fp32 out
  {
    const int waves = (M_ROWS / 32) * (Cq / 64);     // 4096
    proj_gemm<<<waves / 4, 128, 0, stream>>>(AOb, WPRJB, b_proj, out);
  }
}
